// TransformerBlock_11493332484543
// MI455X (gfx1250) — matrix operand order repaired and checked by reading, not compiled, not run on hardware
//
#include <hip/hip_runtime.h>

typedef __attribute__((ext_vector_type(16))) _Float16     v16h;
typedef __attribute__((ext_vector_type(8)))  _Float16     v8h;
typedef __attribute__((ext_vector_type(8)))  float        v8f;
typedef __attribute__((ext_vector_type(8)))  unsigned int v8u;

#define NPIX 16384   // 128*128
#define NB   4

__device__ __forceinline__ float gelu_exact(float v) {
  return 0.5f * v * (1.0f + erff(v * 0.70710678118654752f));
}

// Swizzled f16 activation layout (B-operand friendly):
//   per batch: [pix>>4][c>>5][(c>>4)&1][pix&15][c&15]   (halves)
// Any (pixel, 16-k-group) lives in 16 contiguous halves -> 2 x b128 per B tile.
__device__ __forceinline__ long swz_idx(int c, int pix, int KB) {
  return (((long)(pix >> 4) * KB + (c >> 5)) * 2 + ((c >> 4) & 1)) * 256 +
         (long)(pix & 15) * 16 + (c & 15);
}

// Branch-free zeroing: 1 cndmask + 8 v_and_b32 (folds away when ok is constant)
__device__ __forceinline__ v16h maskv(v16h x, bool ok) {
  unsigned m = ok ? 0xFFFFFFFFu : 0u;
  v8u u = __builtin_bit_cast(v8u, x);
  u &= m;
  return __builtin_bit_cast(v16h, u);
}

union pk32 { _Float16 h[2]; unsigned u; };

// ---------------------------------------------------------------------------
// Weight prep: f32 OIHW -> f16 A-operand layout [tap][Mtot][Kpad], zero padded
// ---------------------------------------------------------------------------
__global__ __launch_bounds__(256)
void prep_weight(const float* __restrict__ W, _Float16* __restrict__ O,
                 int Cout, int Cin, int taps, int Mtot, int Kpad) {
  int i = blockIdx.x * 256 + threadIdx.x;
  int tot = taps * Mtot * Kpad;
  if (i >= tot) return;
  int tap = i / (Mtot * Kpad);
  int rem = i - tap * (Mtot * Kpad);
  int m = rem / Kpad, k = rem - m * Kpad;
  float v = 0.f;
  if (m < Cout && k < Cin) v = W[(m * Cin + k) * taps + tap];
  O[i] = (_Float16)v;
}

// ---------------------------------------------------------------------------
// Generic WMMA implicit-GEMM conv:  Out[m,pix] = sum_k A[m,k] * X[k, pix+shift]
//   BMODE: 0 = swizzled f16 B; 1 = swizzled f16 B * Xmul; 2 = linear f32 B
//   OMODE: 0 = f32 linear; 1 = f16 linear; 2 = f16 swizzled (pair-packed b32)
//   RMODE: 0 = none;       1 = f32 linear resid; 2 = f16 swizzled resid
// ---------------------------------------------------------------------------
template<int COUT, int CIN, int TAPS, int BMODE, int OMODE, int RMODE>
__global__ __launch_bounds__(128)
void wmma_conv_kernel(const void* __restrict__ Xv, int xKB, int xoff0,
                      const void* __restrict__ Mv,
                      const _Float16* __restrict__ A, long ab,
                      const float* __restrict__ bias,
                      const void* __restrict__ Rv, int rKB, int roff0,
                      void* __restrict__ Ov, int oKB, int geluF) {
  constexpr int MT   = (COUT + 15) / 16;
  constexpr int KPAD = (CIN + 31) & ~31;

  const int lane  = threadIdx.x & 31;
  const int wv    = threadIdx.x >> 5;
  const int z     = blockIdx.z;
  const int pt    = blockIdx.x * 4 + wv;
  const int nIdx  = lane & 15;
  const int khalf = (lane >> 4) & 1;
  const int pix0  = pt * 16 + nIdx;
  const int py    = pix0 >> 7, px = pix0 & 127;

  v8f zero = {};
  v8f acc[MT];
#pragma unroll
  for (int t = 0; t < MT; ++t) acc[t] = zero;

  const _Float16* Ab = A + (long)z * ab;
  const _Float16* Xs = nullptr;
  const _Float16* Ms = nullptr;
  const float*    Xf = nullptr;
  if (BMODE == 2) {
    Xf = (const float*)Xv + (long)z * CIN * NPIX;
  } else {
    Xs = (const _Float16*)Xv + (long)z * NPIX * xKB * 32;
    if (BMODE == 1) Ms = (const _Float16*)Mv + (long)z * NPIX * xKB * 32;
  }

#pragma unroll
  for (int tap = 0; tap < TAPS; ++tap) {
    int dy = 0, dx = 0;
    if (TAPS == 9) { dy = tap / 3 - 1; dx = tap % 3 - 1; }
    const int py2 = py + dy, px2 = px + dx;
    const bool valid =
        (TAPS == 1) ? true : (((unsigned)py2 < 128u) && ((unsigned)px2 < 128u));
    const int spix = (TAPS == 1) ? pix0 : (valid ? ((py2 << 7) + px2) : 0);

#pragma unroll
    for (int kb = 0; kb < KPAD; kb += 32) {
      v16h bt;
      if (BMODE == 2) {
        // linear f32 gather + convert (only tiny attention GEMMs use this)
#pragma unroll
        for (int i = 0; i < 16; ++i) {
          const int k = kb + khalf * 16 + i;
          float raw = Xf[(long)k * NPIX + spix];
          bt[i] = (_Float16)((k < CIN) ? raw : 0.f);
        }
      } else {
        const int c0 = xoff0 + kb + khalf * 16;          // 16-aligned
        const long boff = (((long)(spix >> 4) * xKB + (c0 >> 5)) * 2 +
                           ((c0 >> 4) & 1)) * 256 + (long)(spix & 15) * 16;
        v8h b0 = *(const v8h*)(Xs + boff);
        v8h b1 = *(const v8h*)(Xs + boff + 8);
        if (BMODE == 1) {
          v8h m0 = *(const v8h*)(Ms + boff);
          v8h m1 = *(const v8h*)(Ms + boff + 8);
          b0 *= m0;
          b1 *= m1;
        }
        bt = __builtin_shufflevector(b0, b1, 0, 1, 2, 3, 4, 5, 6, 7,
                                     8, 9, 10, 11, 12, 13, 14, 15);
        bt = maskv(bt, valid && ((kb + khalf * 16) < CIN));  // folds when full
      }
      // ---- MT weight tiles, 2 x 16-byte vector loads each ----
#pragma unroll
      for (int t = 0; t < MT; ++t) {
        const v8h* ap = (const v8h*)(Ab +
            ((long)(tap * MT * 16 + t * 16 + nIdx)) * KPAD + kb + khalf * 16);
        v8h a0 = ap[0];
        v8h a1 = ap[1];                                  // +8 halves
        v16h at = __builtin_shufflevector(a0, a1, 0, 1, 2, 3, 4, 5, 6, 7,
                                          8, 9, 10, 11, 12, 13, 14, 15);
        acc[t] = __builtin_amdgcn_wmma_f32_16x16x32_f16(
            false, at, false, bt, (short)0, acc[t], false, false);
      }
    }
  }

  // ---- epilogue: lane n = lane&15, VGPR r -> m = t*16 + r + 8*khalf ----
  if (OMODE == 2) {
    // swizzled f16 out, pair-packed b32 stores (rows r, r+1 are adjacent c)
    _Float16* Oh = (_Float16*)Ov + (long)z * NPIX * oKB * 32;
#pragma unroll
    for (int t = 0; t < MT; ++t) {
#pragma unroll
      for (int r = 0; r < 8; r += 2) {
        const int m = t * 16 + r + 8 * khalf;
        float v0 = acc[t][r], v1 = acc[t][r + 1];
        if (bias)  { v0 += bias[m]; v1 += bias[m + 1]; }
        if (geluF) { v0 = gelu_exact(v0); v1 = gelu_exact(v1); }
        pk32 pk;
        pk.h[0] = (_Float16)v0;
        pk.h[1] = (_Float16)v1;
        *(unsigned*)(Oh + swz_idx(m, pix0, oKB)) = pk.u;
      }
    }
  } else {
    float* Of = (OMODE == 0) ? ((float*)Ov + (long)z * COUT * NPIX) : nullptr;
    _Float16* Oh =
        (OMODE == 1) ? ((_Float16*)Ov + (long)z * COUT * NPIX) : nullptr;
    const float* Rf =
        (RMODE == 1) ? ((const float*)Rv + (long)z * COUT * NPIX) : nullptr;
    const _Float16* Rh =
        (RMODE == 2) ? ((const _Float16*)Rv + (long)z * NPIX * rKB * 32) : nullptr;
#pragma unroll
    for (int t = 0; t < MT; ++t) {
#pragma unroll
      for (int r = 0; r < 8; ++r) {
        const int m = t * 16 + r + 8 * khalf;
        if (m < COUT) {
          float v = acc[t][r];
          if (bias)  v += bias[m];
          if (geluF) v = gelu_exact(v);
          if (RMODE == 1) v += Rf[(long)m * NPIX + pix0];
          if (RMODE == 2) v += (float)Rh[swz_idx(roff0 + m, pix0, rKB)];
          if (OMODE == 0) Of[(long)m * NPIX + pix0] = v;
          else Oh[(long)m * NPIX + pix0] = (_Float16)v;
        }
      }
    }
  }
}

// ---------------------------------------------------------------------------
// Grouped / depthwise 3x3 conv (pad=1), linear-f16 in -> swizzled-f16 out.
// ---------------------------------------------------------------------------
__global__ __launch_bounds__(256)
void grouped_conv3x3(const _Float16* __restrict__ X, int Cin,
                     const float* __restrict__ Wt, const float* __restrict__ bias,
                     _Float16* __restrict__ Out, int oKB,
                     int Cout, int G, int OPG, int geluF) {
  long i = (long)blockIdx.x * 256 + threadIdx.x;
  int z = blockIdx.y;
  int c = (int)(i >> 14);
  long pix = i & (NPIX - 1);
  if (c >= Cout) return;
  int py = (int)(pix >> 7), px = (int)(pix & 127);
  int ic0 = (c / OPG) * G;
  float acc = bias[c];
  for (int j = 0; j < G; ++j) {
    const _Float16* Xc = X + (long)z * Cin * NPIX + (long)(ic0 + j) * NPIX;
#pragma unroll
    for (int t = 0; t < 9; ++t) {
      int dy = t / 3 - 1, dx = t % 3 - 1;
      int py2 = py + dy, px2 = px + dx;
      bool ok = ((unsigned)py2 < 128u) && ((unsigned)px2 < 128u);
      int sp = ok ? ((py2 << 7) + px2) : 0;
      float xv = (float)Xc[sp];
      acc += Wt[(c * G + j) * 9 + t] * (ok ? xv : 0.f);
    }
  }
  if (geluF) acc = gelu_exact(acc);
  Out[(long)z * NPIX * oKB * 32 + swz_idx(c, (int)pix, oKB)] = (_Float16)acc;
}

// ---------------------------------------------------------------------------
// LayerNorm over channel dim; f32 linear in -> swizzled f16 out (b32 pairs).
// ---------------------------------------------------------------------------
__global__ __launch_bounds__(256)
void ln_swz(const float* __restrict__ X, const float* __restrict__ w,
            const float* __restrict__ b, _Float16* __restrict__ Out,
            int C, int oKB) {
  long idx = (long)blockIdx.x * 256 + threadIdx.x;   // over NB*NPIX pixels
  long z = idx >> 14;
  int pix = (int)(idx & (NPIX - 1));
  const float* xp = X + z * (long)C * NPIX + pix;
  float s = 0.f, s2 = 0.f;
  for (int c = 0; c < C; ++c) { float v = xp[(long)c * NPIX]; s += v; s2 += v * v; }
  float mu = s / C;
  float var = s2 / C - mu * mu;
  float rs = rsqrtf(var + 1e-5f);
  _Float16* op = Out + z * (long)NPIX * oKB * 32;
  for (int c = 0; c < C; c += 2) {
    float v0 = (xp[(long)c * NPIX] - mu) * rs * w[c] + b[c];
    float v1 = (xp[(long)(c + 1) * NPIX] - mu) * rs * w[c + 1] + b[c + 1];
    pk32 pk;
    pk.h[0] = (_Float16)v0;
    pk.h[1] = (_Float16)v1;
    *(unsigned*)(op + swz_idx(c, pix, oKB)) = pk.u;
  }
}

// ---------------------------------------------------------------------------
// Row-wise L2 normalize, f32 in-place (attention q/k). One block per row.
// ---------------------------------------------------------------------------
__global__ __launch_bounds__(256)
void l2norm_rows(const float* __restrict__ X, float* __restrict__ Out) {
  __shared__ float red[256];
  const long row = blockIdx.x;
  const float* xp = X + row * (long)NPIX;
  float s = 0.f;
  for (int i = threadIdx.x; i < NPIX; i += 256) { float v = xp[i]; s += v * v; }
  red[threadIdx.x] = s;
  __syncthreads();
  for (int st = 128; st > 0; st >>= 1) {
    if ((int)threadIdx.x < st) red[threadIdx.x] += red[threadIdx.x + st];
    __syncthreads();
  }
  float n = sqrtf(red[0]);
  float sc = 1.0f / fmaxf(n, 1e-12f);
  float* op = Out + row * (long)NPIX;
  for (int i = threadIdx.x; i < NPIX; i += 256) op[i] = xp[i] * sc;
}

// ---------------------------------------------------------------------------
// 24x24 gram via WMMA, K split over blocks, f32 atomic reduce.
// ---------------------------------------------------------------------------
__global__ __launch_bounds__(32)
void wmma_gram_kernel(const float* __restrict__ Q, const float* __restrict__ Kt,
                      float* __restrict__ AttnOut, int kchunk) {
  const int lane  = threadIdx.x & 31;
  const int bh    = blockIdx.y;
  const int nIdx  = lane & 15;
  const int khalf = (lane >> 4) & 1;
  const int kstart = blockIdx.x * kchunk;
  const float* Qb = Q + (long)bh * 24 * NPIX;
  const float* Kb = Kt + (long)bh * 24 * NPIX;
  v8f zero = {};
  v8f acc[2][2];
  acc[0][0] = zero; acc[0][1] = zero; acc[1][0] = zero; acc[1][1] = zero;

  for (int kb = kstart; kb < kstart + kchunk; kb += 32) {
    v16h at[2], bt[2];
#pragma unroll
    for (int mt = 0; mt < 2; ++mt) {
      const int c = mt * 16 + nIdx;
      const float* qr = Qb + (long)c * NPIX + kb;
#pragma unroll
      for (int j = 0; j < 8; ++j) {
        at[mt][j]     = (_Float16)qr[khalf * 16 + j];
        at[mt][8 + j] = (_Float16)qr[khalf * 16 + 8 + j];
      }
      at[mt] = maskv(at[mt], (mt == 0) ? true : (c < 24));
    }
#pragma unroll
    for (int nt = 0; nt < 2; ++nt) {
      const int d = nt * 16 + nIdx;
      const float* kr = Kb + (long)d * NPIX + kb + khalf * 16;
#pragma unroll
      for (int i = 0; i < 16; ++i) bt[nt][i] = (_Float16)kr[i];
      bt[nt] = maskv(bt[nt], (nt == 0) ? true : (d < 24));
    }
#pragma unroll
    for (int mt = 0; mt < 2; ++mt)
#pragma unroll
      for (int nt = 0; nt < 2; ++nt)
        acc[mt][nt] = __builtin_amdgcn_wmma_f32_16x16x32_f16(
            false, at[mt], false, bt[nt], (short)0, acc[mt][nt], false, false);
  }
#pragma unroll
  for (int mt = 0; mt < 2; ++mt)
#pragma unroll
    for (int nt = 0; nt < 2; ++nt)
#pragma unroll
      for (int r = 0; r < 8; ++r) {
        const int c = mt * 16 + r + 8 * khalf;
        const int d = nt * 16 + nIdx;
        if (c < 24 && d < 24)
          atomicAdd(&AttnOut[bh * 576 + c * 24 + d], acc[mt][nt][r]);
      }
}

__global__ __launch_bounds__(64)
void softmax24(float* __restrict__ A, const float* __restrict__ temp) {
  int r = blockIdx.x * 64 + threadIdx.x;
  if (r >= 8 * 24) return;
  int bh = r / 24;
  int h = bh & 1;
  float* p = A + bh * 576 + (r % 24) * 24;
  float t = temp[h];
  float v[24];
  float mx = -3.4e38f;
#pragma unroll
  for (int i = 0; i < 24; ++i) { v[i] = p[i] * t; mx = fmaxf(mx, v[i]); }
  float s = 0.f;
#pragma unroll
  for (int i = 0; i < 24; ++i) { v[i] = expf(v[i] - mx); s += v[i]; }
  float inv = 1.f / s;
#pragma unroll
  for (int i = 0; i < 24; ++i) p[i] = v[i] * inv;
}

// Pack softmaxed attn (f32 [bh][24][24]) into f16 A layout [bh][32][32], padded
__global__ __launch_bounds__(256)
void cvt_attn(const float* __restrict__ A, _Float16* __restrict__ O) {
  int i = blockIdx.x * 256 + threadIdx.x;
  if (i >= 8 * 1024) return;
  int bh = i >> 10;
  int m = (i >> 5) & 31;
  int k = i & 31;
  float v = (m < 24 && k < 24) ? A[bh * 576 + m * 24 + k] : 0.f;
  O[i] = (_Float16)v;
}

// BiBranch gating: xa = x*g1 + pc ; pa = pc*g2 + xa.
// x, pc swizzled-f16 in; g linear-f16 in; concat(xa,pa) -> swizzled (KB=3).
__global__ __launch_bounds__(256)
void ew_gate(const _Float16* __restrict__ Xs, int xKB, int xoff0,
             const _Float16* __restrict__ G96, const _Float16* __restrict__ PCs,
             _Float16* __restrict__ O) {
  long i = (long)blockIdx.x * 256 + threadIdx.x;    // over 48*NPIX
  int z = blockIdx.y;
  int c = (int)(i >> 14);
  int pix = (int)(i & (NPIX - 1));
  float g1 = (float)G96[(long)z * 96 * NPIX + (long)c * NPIX + pix];
  float g2 = (float)G96[(long)z * 96 * NPIX + (long)(c + 48) * NPIX + pix];
  float xv = (float)Xs[(long)z * NPIX * xKB * 32 + swz_idx(xoff0 + c, pix, xKB)];
  float pv = (float)PCs[(long)z * NPIX * 2 * 32 + swz_idx(c, pix, 2)];
  float xa = xv * g1 + pv;
  float pa = pv * g2 + xa;
  _Float16* oz = O + (long)z * NPIX * 3 * 32;
  oz[swz_idx(c, pix, 3)]      = (_Float16)xa;
  oz[swz_idx(c + 48, pix, 3)] = (_Float16)pa;
}

__global__ __launch_bounds__(256)
void fill0(float* __restrict__ p, int n) {
  int i = blockIdx.x * 256 + threadIdx.x;
  if (i < n) p[i] = 0.f;
}

// ---------------------------------------------------------------------------
// Host-side launch helpers
// ---------------------------------------------------------------------------
template<int COUT, int CIN, int TAPS, int BMODE, int OMODE, int RMODE>
static void launch_gemm(hipStream_t s, const void* X, int xKB, int xoff0,
                        const void* M, const _Float16* A, long ab,
                        const float* bias, const void* R, int rKB, int roff0,
                        void* Out, int oKB, int Z, int geluF) {
  dim3 grid(NPIX / 64, 1, Z), blk(128);
  wmma_conv_kernel<COUT, CIN, TAPS, BMODE, OMODE, RMODE><<<grid, blk, 0, s>>>(
      X, xKB, xoff0, M, A, ab, bias, R, rKB, roff0, Out, oKB, geluF);
}

struct MTW {
  const _Float16 *perc, *fuse1, *gate, *gmu, *dw1, *dw3, *c12;
  const float *perc_b, *fuse1_b, *gate_b, *gmu_b, *dw1_b, *dw2_w, *dw2_b, *dw3_b, *c12_b;
};

extern "C" void kernel_launch(void* const* d_in, const int* in_sizes, int n_in,
                              void* d_out, int out_size, void* d_ws, size_t ws_size,
                              hipStream_t stream) {
  (void)in_sizes; (void)n_in; (void)out_size; (void)ws_size;
  auto in = [&](int i) { return (const float*)d_in[i]; };

  char* ws = (char*)d_ws;
  size_t off = 0;
  auto allocF = [&](size_t elems) {
    float* p = (float*)(ws + off);
    off = (off + elems * 4 + 255) & ~(size_t)255;
    return p;
  };
  auto allocH = [&](size_t elems) {
    _Float16* p = (_Float16*)(ws + off);
    off = (off + elems * 2 + 255) & ~(size_t)255;
    return p;
  };
  auto allocS = [&](int KB) {              // swizzled f16 tensor
    return allocH((size_t)NB * NPIX * KB * 32);
  };

  const size_t CH = (size_t)NB * NPIX;

  // f32 buffers (adjacent: padded-K over-reads of the f32-B GEMMs stay in ws)
  float* qm   = allocF(48 * CH);
  float* km   = allocF(48 * CH);
  float* vm   = allocF(48 * CH);   // attn@v over-reads tail -> aout
  float* aout = allocF(48 * CH);   // proj over-reads tail -> xAtt
  float* xAtt = allocF(48 * CH);
  float* attnBuf = allocF(8 * 576);

  // swizzled f16 activations
  _Float16* pLN  = allocS(5);      // 144ch (pad to 160)
  _Float16* xn   = allocS(2);      // 48ch (pad 64)
  _Float16* bufB = allocS(5);      // qkv-dw out, 144ch
  _Float16* pc   = allocS(2);
  _Float16* u    = allocS(2);
  _Float16* vv   = allocS(2);
  _Float16* f1   = allocS(2);
  _Float16* pi   = allocS(2);
  _Float16* s96b = allocS(3);      // concat(xa,pa) / dw2-out / pout1-out

  // linear f16 activations (grouped-conv inputs)
  _Float16* bufA  = allocH(144 * CH);
  _Float16* s96a  = allocH(96 * CH);
  _Float16* mtout = allocH(48 * CH);
  _Float16* attn16 = allocH(8 * 1024);

  // --- prep all 1x1 / dense-3x3 weights into f16 WMMA-A layout ---
  auto prep = [&](int idx, _Float16* dst, int Cout, int Cin, int taps) {
    int Mtot = ((Cout + 15) >> 4) << 4;
    int Kpad = (Cin + 31) & ~31;
    int n = taps * Mtot * Kpad;
    prep_weight<<<dim3((n + 255) / 256), 256, 0, stream>>>(in(idx), dst, Cout, Cin,
                                                           taps, Mtot, Kpad);
  };

  _Float16* qkv16 = allocH(144 * 64);  prep(8, qkv16, 144, 48, 1);
  _Float16* proj16 = allocH(48 * 64);  prep(29, proj16, 48, 48, 1);
  _Float16* pin116 = allocH(96 * 64);  prep(31, pin116, 96, 48, 1);
  _Float16* pout216 = allocH(48 * 96); prep(53, pout216, 48, 96, 1);

  auto makeMT = [&](int b0) {
    MTW w;
    _Float16* t;
    t = allocH(48 * 160);    prep(b0 + 0, t, 48, 144, 1); w.perc = t;  w.perc_b = in(b0 + 1);
    t = allocH(48 * 64);     prep(b0 + 2, t, 48, 48, 1);  w.fuse1 = t; w.fuse1_b = in(b0 + 3);
    t = allocH(9 * 96 * 64); prep(b0 + 4, t, 96, 48, 9);  w.gate = t;  w.gate_b = in(b0 + 5);
    t = allocH(48 * 96);     prep(b0 + 6, t, 48, 96, 1);  w.gmu = t;   w.gmu_b = in(b0 + 7);
    t = allocH(96 * 64);     prep(b0 + 8, t, 96, 48, 1);  w.dw1 = t;   w.dw1_b = in(b0 + 9);
    w.dw2_w = in(b0 + 10);   w.dw2_b = in(b0 + 11);
    t = allocH(48 * 96);     prep(b0 + 12, t, 48, 96, 1); w.dw3 = t;   w.dw3_b = in(b0 + 13);
    t = allocH(48 * 64);     prep(b0 + 14, t, 48, 48, 1); w.c12 = t;   w.c12_b = in(b0 + 15);
    return w;
  };
  MTW amt = makeMT(13);
  MTW fmt = makeMT(35);

  auto ln = [&](const float* X, const float* w, const float* b, _Float16* O,
                int C, int oKB) {
    ln_swz<<<dim3(NB * NPIX / 256), 256, 0, stream>>>(X, w, b, O, C, oKB);
  };
  auto dwconv = [&](const _Float16* X, int Cin, const float* W, const float* b,
                    _Float16* O, int oKB, int Cout, int G, int OPG, int gl) {
    dim3 g((unsigned)((long)Cout * NPIX / 256), NB);
    grouped_conv3x3<<<g, 256, 0, stream>>>(X, Cin, W, b, O, oKB, Cout, G, OPG, gl);
  };

  // Matching transformation (attention q/k/v and FFN).  x: swizzled f16.
  auto mt_forward = [&](const _Float16* x, int xKB, int xo, const MTW& w,
                        void* out, bool outF32) {
    launch_gemm<48, 144, 1, 0, 2, 0>(stream, pLN, 5, 0, nullptr, w.perc, 0,
                                     w.perc_b, nullptr, 0, 0, pc, 2, NB, 0);
    launch_gemm<48, 48, 1, 0, 2, 0>(stream, x, xKB, xo, nullptr, w.fuse1, 0,
                                    w.fuse1_b, nullptr, 0, 0, u, 2, NB, 0);
    launch_gemm<48, 48, 1, 0, 2, 0>(stream, pc, 2, 0, nullptr, w.fuse1, 0,
                                    w.fuse1_b, nullptr, 0, 0, vv, 2, NB, 0);
    // dense 3x3 gate over (u * vv), GELU epilogue -> linear f16
    launch_gemm<96, 48, 9, 1, 1, 0>(stream, u, 2, 0, vv, w.gate, 0, w.gate_b,
                                    nullptr, 0, 0, s96a, 0, NB, 1);
    ew_gate<<<dim3(48 * NPIX / 256, NB), 256, 0, stream>>>(x, xKB, xo, s96a, pc,
                                                           s96b);
    launch_gemm<48, 96, 1, 0, 2, 0>(stream, s96b, 3, 0, nullptr, w.gmu, 0,
                                    w.gmu_b, nullptr, 0, 0, f1, 2, NB, 0);
    launch_gemm<96, 48, 1, 0, 1, 0>(stream, f1, 2, 0, nullptr, w.dw1, 0,
                                    w.dw1_b, nullptr, 0, 0, s96a, 0, NB, 0);
    dwconv(s96a, 96, w.dw2_w, w.dw2_b, s96b, 3, 96, 1, 1, 1);  // depthwise+GELU
    launch_gemm<48, 96, 1, 0, 2, 0>(stream, s96b, 3, 0, nullptr, w.dw3, 0,
                                    w.dw3_b, nullptr, 0, 0, u, 2, NB, 0);
    // c12 @ (dd * f1) + x
    if (outF32)
      launch_gemm<48, 48, 1, 1, 0, 2>(stream, u, 2, 0, f1, w.c12, 0, w.c12_b,
                                      x, xKB, xo, out, 0, NB, 0);
    else
      launch_gemm<48, 48, 1, 1, 1, 2>(stream, u, 2, 0, f1, w.c12, 0, w.c12_b,
                                      x, xKB, xo, out, 0, NB, 0);
  };

  // ---------------- forward ----------------
  ln(in(1), in(2), in(3), pLN, 144, 5);          // p = LN(perception)
  ln(in(0), in(4), in(5), xn, 48, 2);            // LN(x) for attention

  launch_gemm<144, 48, 1, 0, 1, 0>(stream, xn, 2, 0, nullptr, qkv16, 0, in(9),
                                   nullptr, 0, 0, bufA, 0, NB, 0);   // qkv 1x1
  dwconv(bufA, 144, in(10), in(11), bufB, 5, 144, 1, 1, 0);          // qkv dw 3x3

  mt_forward(bufB, 5, 0, amt, qm, true);
  mt_forward(bufB, 5, 48, amt, km, true);
  mt_forward(bufB, 5, 96, amt, vm, true);

  l2norm_rows<<<dim3(NB * 48), 256, 0, stream>>>(qm, qm);
  l2norm_rows<<<dim3(NB * 48), 256, 0, stream>>>(km, km);

  fill0<<<dim3((8 * 576 + 255) / 256), 256, 0, stream>>>(attnBuf, 8 * 576);
  wmma_gram_kernel<<<dim3(64, 8), 32, 0, stream>>>(qm, km, attnBuf, NPIX / 64);
  softmax24<<<dim3(3), 64, 0, stream>>>(attnBuf, in(12));
  cvt_attn<<<dim3(8 * 1024 / 256), 256, 0, stream>>>(attnBuf, attn16);

  // out = attn @ v (z = b*h, per-z A matrix), linear-f32 B
  launch_gemm<24, 24, 1, 2, 0, 0>(stream, vm, 0, 0, nullptr, attn16, 1024,
                                  nullptr, nullptr, 0, 0, aout, 0, 8, 0);
  // x = x + proj(out)
  launch_gemm<48, 48, 1, 2, 0, 1>(stream, aout, 0, 0, nullptr, proj16, 0,
                                  in(30), in(0), 0, 0, xAtt, 0, NB, 0);

  // ---------------- FFN ----------------
  ln(xAtt, in(6), in(7), xn, 48, 2);
  launch_gemm<96, 48, 1, 0, 1, 0>(stream, xn, 2, 0, nullptr, pin116, 0, in(32),
                                  nullptr, 0, 0, s96a, 0, NB, 0);    // pin1
  dwconv(s96a, 96, in(33), in(34), pi, 2, 48, 2, 1, 0);              // pin2
  mt_forward(pi, 2, 0, fmt, mtout, false);                           // -> lin f16
  dwconv(mtout, 48, in(51), in(52), s96b, 3, 96, 1, 2, 0);           // pout1
  // final: d_out = xAtt + pout2 @ pout1_out
  launch_gemm<48, 96, 1, 0, 0, 1>(stream, s96b, 3, 0, nullptr, pout216, 0,
                                  in(54), xAtt, 0, 0, d_out, 0, NB, 0);
}
